// MultiHeadSelfAttentionV2_10179072491709
// MI455X (gfx1250) — matrix operand order repaired and checked by reading, not compiled, not run on hardware
//
#include <hip/hip_runtime.h>
#include <hip/hip_bf16.h>

#define EMB   1024
#define HEADS 16
#define HD    64
#define BATCH 4
#define SEQ   2048
#define MROWS (BATCH*SEQ)   // 8192 token rows

typedef __bf16 bf16_t;
typedef __attribute__((ext_vector_type(16))) __bf16 v16bf;
typedef __attribute__((ext_vector_type(8)))  __bf16 v8bf;
typedef __attribute__((ext_vector_type(8)))  float  v8f;

__device__ inline v8f vzero8() {
  v8f r;
#pragma unroll
  for (int i = 0; i < 8; ++i) r[i] = 0.f;
  return r;
}

// Wave-local LDS ordering: DS pipe is in-order per wave; we only need to stop
// the compiler from reordering and to drain DScnt before dependent reads.
__device__ inline void lds_fence() {
  __builtin_amdgcn_wave_barrier();
  asm volatile("s_wait_dscnt 0" ::: "memory");
  __builtin_amdgcn_wave_barrier();
}

// ---------------------------------------------------------------------------
// fp32 -> bf16 conversion (grid-stride)
// ---------------------------------------------------------------------------
__global__ __launch_bounds__(256)
void cvt_bf16(const float* __restrict__ in, bf16_t* __restrict__ out, int n) {
  int i = blockIdx.x * blockDim.x + threadIdx.x;
  int stride = gridDim.x * blockDim.x;
  for (; i < n; i += stride) out[i] = (bf16_t)in[i];
}

// ---------------------------------------------------------------------------
// Y[m,f] = sum_e A[m,e]*W[f,e] + bias[f]   (torch Linear: x @ W.T + b)
// A: [MROWS x EMB] bf16 row-major, W: [EMB x EMB] bf16 row-major.
// Register-blocked 32x64 tile per wave: 2 M x 4 N 16x16 sub-tiles,
// 8 v_wmma per 32-deep K step; A fragments reused 4x, B fragments 2x.
// mode 0: store bf16 to [b,h,t,d] (Q/K)   mode 1: store bf16 to [b,h,d,t] (V^T)
// mode 2: store fp32 to [m,f]             (projection)
// ---------------------------------------------------------------------------
__global__ __launch_bounds__(256)
void gemm_wmma_blk(const bf16_t* __restrict__ A, const bf16_t* __restrict__ W,
                   const float* __restrict__ bias, void* __restrict__ outp,
                   int mode) {
  const int wave = threadIdx.x >> 5;
  const int lane = threadIdx.x & 31;
  const int tile = blockIdx.x * 8 + wave;       // 256 M-blk x 16 N-blk = 4096
  const int tn = tile & 15;                     // N block of 64
  const int tm = tile >> 4;                     // M block of 32
  const int m0 = tm << 5, n0 = tn << 6;
  const int half = lane >> 4;
  const int l16  = lane & 15;

  const bf16_t* arow0 = A + (size_t)(m0 + l16) * EMB;
  const bf16_t* arow1 = arow0 + (size_t)16 * EMB;
  const bf16_t* brow0 = W + (size_t)(n0 + l16) * EMB;

  v8f acc[2][4];
#pragma unroll
  for (int mi = 0; mi < 2; ++mi)
#pragma unroll
    for (int j = 0; j < 4; ++j) acc[mi][j] = vzero8();

#pragma unroll 2
  for (int kk = 0; kk < EMB; kk += 32) {
    // prefetch ~1KB ahead on each stream (global_prefetch_b8; OOB is dropped)
    __builtin_prefetch(arow0 + kk + 512, 0, 1);
    __builtin_prefetch(arow1 + kk + 512, 0, 1);
    __builtin_prefetch(brow0 + kk + 512, 0, 1);

    // A operands: lane half 0 holds K 0..15, half 1 16..31 (contiguous)
    v16bf av[2];
#pragma unroll
    for (int mi = 0; mi < 2; ++mi) {
      const bf16_t* ar = mi ? arow1 : arow0;
      v8bf alo = *(const v8bf*)(ar + kk + half * 16);
      v8bf ahi = *(const v8bf*)(ar + kk + half * 16 + 8);
#pragma unroll
      for (int j = 0; j < 8; ++j) { av[mi][j] = alo[j]; av[mi][j + 8] = ahi[j]; }
    }
    // B operands: lane = column n; halves hold K 0..15 / 16..31 (contiguous)
    v16bf bv[4];
#pragma unroll
    for (int j = 0; j < 4; ++j)
      bv[j] = *(const v16bf*)(brow0 + (size_t)(j * 16) * EMB + kk + half * 16);

#pragma unroll
    for (int mi = 0; mi < 2; ++mi)
#pragma unroll
      for (int j = 0; j < 4; ++j)
        acc[mi][j] = __builtin_amdgcn_wmma_f32_16x16x32_bf16(
            false, av[mi], false, bv[j], (short)0, acc[mi][j], false, false);
  }

#pragma unroll
  for (int j = 0; j < 4; ++j) {
    const int f = n0 + j * 16 + l16;
    const float bcol = bias[f];
    if (mode == 2) {
      float* O = (float*)outp;
#pragma unroll
      for (int mi = 0; mi < 2; ++mi)
#pragma unroll
        for (int i = 0; i < 8; ++i) {
          int r = m0 + mi * 16 + i + half * 8;   // C layout row mapping
          O[(size_t)r * EMB + f] = acc[mi][j][i] + bcol;
        }
    } else {
      bf16_t* O = (bf16_t*)outp;
      const int h = f >> 6, d = f & 63;
#pragma unroll
      for (int mi = 0; mi < 2; ++mi)
#pragma unroll
        for (int i = 0; i < 8; ++i) {
          int m = m0 + mi * 16 + i + half * 8;
          int b = m >> 11, t = m & (SEQ - 1);
          size_t idx = (mode == 0)
              ? (((size_t)(b * HEADS + h) * SEQ + t) * HD + d)   // [b,h,t,d]
              : (((size_t)(b * HEADS + h) * HD + d) * SEQ + t);  // [b,h,d,t]
          O[idx] = (bf16_t)(acc[mi][j][i] + bcol);
        }
    }
  }
}

// ---------------------------------------------------------------------------
// Flash attention: one wave per (b,h,q-tile of 16). Causal, scale=1/sqrt(T).
// Writes Ob = bf16(attn_out + residual x), layout [b,t,e].
// ---------------------------------------------------------------------------
__global__ __launch_bounds__(256)
void attn_wmma_k(const bf16_t* __restrict__ Qb, const bf16_t* __restrict__ Kb,
                 const bf16_t* __restrict__ Vt, const float* __restrict__ x,
                 bf16_t* __restrict__ Ob) {
  __shared__ float pbuf[8][16 * 32];              // per-wave P staging slab
  const int wave = threadIdx.x >> 5;
  const int lane = threadIdx.x & 31;
  const int gw = blockIdx.x * 8 + wave;           // 8192 waves total
  const int qt = gw & 127;                        // q tile (T/16 = 128)
  const int bh = gw >> 7;                         // b*H + h
  const int half = lane >> 4;
  const int l16  = lane & 15;

  const bf16_t* Qh = Qb + (size_t)bh * SEQ * HD;
  const bf16_t* Kh = Kb + (size_t)bh * SEQ * HD;
  const bf16_t* Vh = Vt + (size_t)bh * HD * SEQ;
  const int m0 = qt << 4;

  // Q A-operands for K-dim = hd = 64 -> two 16x32 chunks, kept in regs
  v16bf aq[2];
  {
    const bf16_t* qrow = Qh + (size_t)(m0 + l16) * HD;
#pragma unroll
    for (int c = 0; c < 2; ++c) {
      v8bf lo = *(const v8bf*)(qrow + c * 32 + half * 16);
      v8bf hi = *(const v8bf*)(qrow + c * 32 + half * 16 + 8);
#pragma unroll
      for (int j = 0; j < 8; ++j) { aq[c][j] = lo[j]; aq[c][j + 8] = hi[j]; }
    }
  }

  v8f oacc[4];
#pragma unroll
  for (int c = 0; c < 4; ++c) oacc[c] = vzero8();
  v8f mrow, lrow;
#pragma unroll
  for (int i = 0; i < 8; ++i) { mrow[i] = -3.0e38f; lrow[i] = 0.f; }

  const float scale = 0.022097086912079612f;      // 1/sqrt(2048)
  const int qbase = m0 + half * 8;                // C-layout row of element i
  const int nk32 = (qt >> 1) + 1;                 // 32-key steps (mask covers pad)
  float* P = pbuf[wave];

  for (int it = 0; it < nk32; ++it) {
    const int k0 = it * 32;
    // prefetch next key/value tiles (dropped if OOB)
    __builtin_prefetch(Kh + (size_t)(k0 + 32 + l16) * HD, 0, 1);
    __builtin_prefetch(Vh + (size_t)l16 * SEQ + k0 + 32, 0, 1);

    v8f s0, s1;
#pragma unroll
    for (int st = 0; st < 2; ++st) {
      const int kb = k0 + st * 16;
      const bf16_t* krow = Kh + (size_t)(kb + l16) * HD;   // B col n = key kb+l16
      v8f sc = vzero8();
#pragma unroll
      for (int c = 0; c < 2; ++c) {
        v16bf bk = *(const v16bf*)(krow + c * 32 + half * 16);
        sc = __builtin_amdgcn_wmma_f32_16x16x32_bf16(false, aq[c], false, bk,
                                                     (short)0, sc, false, false);
      }
      const int kcol = kb + l16;
      v8f& s = st ? s1 : s0;
#pragma unroll
      for (int i = 0; i < 8; ++i) {
        float v = sc[i] * scale;
        s[i] = (kcol <= qbase + i) ? v : -3.0e38f;         // causal mask
      }
    }

    // --- online softmax; row reductions = xor-shuffles within 16-lane halves,
    //     exactly matching the C-matrix row/lane striping -------------------
    v8f t;
#pragma unroll
    for (int i = 0; i < 8; ++i) t[i] = fmaxf(s0[i], s1[i]);
#pragma unroll
    for (int off = 1; off < 16; off <<= 1)
#pragma unroll
      for (int i = 0; i < 8; ++i) t[i] = fmaxf(t[i], __shfl_xor(t[i], off, 32));

    v8f mnew, alpha, rsum;
#pragma unroll
    for (int i = 0; i < 8; ++i) {
      mnew[i]  = fmaxf(mrow[i], t[i]);
      alpha[i] = __expf(mrow[i] - mnew[i]);
      float p0 = __expf(s0[i] - mnew[i]);
      float p1 = __expf(s1[i] - mnew[i]);
      s0[i] = p0; s1[i] = p1;
      rsum[i] = p0 + p1;
    }
#pragma unroll
    for (int off = 1; off < 16; off <<= 1)
#pragma unroll
      for (int i = 0; i < 8; ++i) rsum[i] += __shfl_xor(rsum[i], off, 32);
#pragma unroll
    for (int i = 0; i < 8; ++i) {
      lrow[i] = lrow[i] * alpha[i] + rsum[i];
      mrow[i] = mnew[i];
    }
#pragma unroll
    for (int c = 0; c < 4; ++c)
#pragma unroll
      for (int i = 0; i < 8; ++i) oacc[c][i] *= alpha[i];

    // --- P: C layout -> A layout via per-wave LDS slab --------------------
#pragma unroll
    for (int i = 0; i < 8; ++i) {
      P[(i + half * 8) * 32 + l16]      = s0[i];
      P[(i + half * 8) * 32 + 16 + l16] = s1[i];
    }
    lds_fence();
    v16bf pa;
    {
      const float* prow = P + l16 * 32;           // A row = l16
#pragma unroll
      for (int j = 0; j < 8; ++j) {
        pa[j]     = (bf16_t)prow[half * 16 + j];          // K {0..7}/{16..23}
        pa[j + 8] = (bf16_t)prow[half * 16 + 8 + j];      // K {8..15}/{24..31}
      }
    }
    lds_fence();

    // --- O += P(16x32) @ V(32x64): 4 WMMAs, V^T rows are contiguous -------
#pragma unroll
    for (int c = 0; c < 4; ++c) {
      const bf16_t* vrow = Vh + (size_t)(c * 16 + l16) * SEQ + k0 + half * 16;
      v16bf bvv = *(const v16bf*)vrow;
      oacc[c] = __builtin_amdgcn_wmma_f32_16x16x32_bf16(false, pa, false, bvv,
                                                        (short)0, oacc[c],
                                                        false, false);
    }
  }

  // epilogue: normalize, add residual, emit bf16 for projection GEMM
  const int h = bh & (HEADS - 1);
  const int b = bh >> 4;
#pragma unroll
  for (int c = 0; c < 4; ++c) {
    const int d = c * 16 + l16;
#pragma unroll
    for (int i = 0; i < 8; ++i) {
      const int tq = m0 + i + half * 8;
      const size_t e = ((size_t)b * SEQ + tq) * EMB + h * HD + d;
      float v = oacc[c][i] / lrow[i];
      Ob[e] = (bf16_t)(v + x[e]);
    }
  }
}

// ---------------------------------------------------------------------------
// LayerNorm over last dim (1024), one block per row, eps = 1e-6
// ---------------------------------------------------------------------------
__global__ __launch_bounds__(256)
void layernorm_k(const float* __restrict__ in, const float* __restrict__ gamma,
                 const float* __restrict__ beta, float* __restrict__ out) {
  __shared__ float s1[256], s2[256];
  const int row = blockIdx.x;
  const float* r = in + (size_t)row * EMB;
  float s = 0.f, ss = 0.f;
  for (int c = threadIdx.x; c < EMB; c += 256) {
    float v = r[c];
    s += v; ss += v * v;
  }
  s1[threadIdx.x] = s; s2[threadIdx.x] = ss;
  __syncthreads();
  for (int off = 128; off > 0; off >>= 1) {
    if ((int)threadIdx.x < off) {
      s1[threadIdx.x] += s1[threadIdx.x + off];
      s2[threadIdx.x] += s2[threadIdx.x + off];
    }
    __syncthreads();
  }
  const float mu  = s1[0] * (1.f / EMB);
  const float var = s2[0] * (1.f / EMB) - mu * mu;
  const float inv = rsqrtf(var + 1e-6f);
  float* o = out + (size_t)row * EMB;
  for (int c = threadIdx.x; c < EMB; c += 256)
    o[c] = (r[c] - mu) * inv * gamma[c] + beta[c];
}

// ---------------------------------------------------------------------------
extern "C" void kernel_launch(void* const* d_in, const int* in_sizes, int n_in,
                              void* d_out, int out_size, void* d_ws,
                              size_t ws_size, hipStream_t stream) {
  const float* x     = (const float*)d_in[0];
  const float* Wq    = (const float*)d_in[1];
  const float* bq    = (const float*)d_in[2];
  const float* Wk    = (const float*)d_in[3];
  const float* bk    = (const float*)d_in[4];
  const float* Wv    = (const float*)d_in[5];
  const float* bv    = (const float*)d_in[6];
  const float* Wp    = (const float*)d_in[7];
  const float* bp    = (const float*)d_in[8];
  const float* gamma = (const float*)d_in[9];
  const float* beta  = (const float*)d_in[10];
  float* out = (float*)d_out;

  char* ws = (char*)d_ws;
  const size_t MB = 1024 * 1024;
  bf16_t* Xb  = (bf16_t*)(ws + 0);        // 16 MB  x bf16 [m,e]
  bf16_t* Wqb = (bf16_t*)(ws + 16 * MB);  //  2 MB
  bf16_t* Wkb = (bf16_t*)(ws + 18 * MB);  //  2 MB
  bf16_t* Wvb = (bf16_t*)(ws + 20 * MB);  //  2 MB
  bf16_t* Wpb = (bf16_t*)(ws + 22 * MB);  //  2 MB
  bf16_t* Qb  = (bf16_t*)(ws + 24 * MB);  // 16 MB  [b,h,t,d]
  bf16_t* Kb  = (bf16_t*)(ws + 40 * MB);  // 16 MB  [b,h,t,d]
  bf16_t* Vt  = (bf16_t*)(ws + 56 * MB);  // 16 MB  [b,h,d,t]
  bf16_t* Ob  = (bf16_t*)(ws + 72 * MB);  // 16 MB  attn_out + residual, bf16
  float*  P2  = (float*)(ws + 88 * MB);   // 32 MB  pre-LN fp32

  cvt_bf16<<<4096, 256, 0, stream>>>(x,  Xb,  MROWS * EMB);
  cvt_bf16<<<1024, 256, 0, stream>>>(Wq, Wqb, EMB * EMB);
  cvt_bf16<<<1024, 256, 0, stream>>>(Wk, Wkb, EMB * EMB);
  cvt_bf16<<<1024, 256, 0, stream>>>(Wv, Wvb, EMB * EMB);
  cvt_bf16<<<1024, 256, 0, stream>>>(Wp, Wpb, EMB * EMB);

  gemm_wmma_blk<<<512, 256, 0, stream>>>(Xb, Wqb, bq, Qb, 0);
  gemm_wmma_blk<<<512, 256, 0, stream>>>(Xb, Wkb, bk, Kb, 0);
  gemm_wmma_blk<<<512, 256, 0, stream>>>(Xb, Wvb, bv, Vt, 1);

  attn_wmma_k<<<1024, 256, 0, stream>>>(Qb, Kb, Vt, x, Ob);

  gemm_wmma_blk<<<512, 256, 0, stream>>>(Ob, Wpb, bp, P2, 2);

  layernorm_k<<<MROWS, 256, 0, stream>>>(P2, gamma, beta, out);
}
